// DeeperAttentionGCNSW_20005957665498
// MI455X (gfx1250) — matrix operand order repaired and checked by reading, not compiled, not run on hardware
//
#include <hip/hip_runtime.h>
#include <hip/hip_bf16.h>

// ---------------- problem constants (match reference) ----------------
#define N_NODES 20000
#define E_EDGES 320000
#define T_STEPS 16
#define F_INP   128
#define HC      256          // H1*F1 == H2*F2 (4 heads x 64)
#define FH      64           // per-head channels == F2 == GRU hidden
#define E_TOT   (E_EDGES + N_NODES)
#define NEG_SLOPE 0.2f
#define EPS_DENOM 1e-16f

typedef __attribute__((ext_vector_type(16))) _Float16 v16h;
typedef __attribute__((ext_vector_type(8)))  float    v8f;

// ---------------- small helpers ----------------
__device__ __forceinline__ void atomicMaxFloat(float* addr, float val) {
    // works with buffer initialized to -inf (0xFF800000)
    if (val >= 0.0f) atomicMax((int*)addr, __float_as_int(val));
    else             atomicMin((unsigned int*)addr, __float_as_uint(val));
}

__device__ __forceinline__ void edge_sd(const int* __restrict__ ei, int e, int& s, int& d) {
    if (e < E_EDGES) { s = ei[e]; d = ei[E_EDGES + e]; }
    else             { s = e - E_EDGES; d = s; }            // self loops
}

__device__ __forceinline__ float sigmoidf_(float x) { return 1.0f / (1.0f + expf(-x)); }

// ---------------- utility kernels ----------------
__global__ void k_fill_f32(float* __restrict__ p, float v, long n) {
    long i = (long)blockIdx.x * blockDim.x + threadIdx.x;
    if (i < n) p[i] = v;
}

// convert + transpose: dst[n*K + k] = (f16) src[k*NCOL + n]   (src is [K][NCOL])
__global__ void k_cvt_f16_T(const float* __restrict__ src, _Float16* __restrict__ dst,
                            int K, int NCOL) {
    int i = blockIdx.x * blockDim.x + threadIdx.x;
    if (i >= K * NCOL) return;
    int n = i / K, k = i - n * K;
    dst[i] = (_Float16)src[(long)k * NCOL + n];
}

// ---------------- WMMA GEMM: C[M,N] = A[M,K] (f32, strided) * Bt (f16) (+bias) ----
// Bt is the weight matrix pre-transposed: Bt[n*K + k]  (so staging is contiguous)
// block = 256 threads = 8 waves; block tile = 128 rows x (NT*16) cols; K step = 32
template <int NT>
__global__ __launch_bounds__(256)
void k_gemm_wmma(const float* __restrict__ A, long strideA,
                 const _Float16* __restrict__ Bt,     // [NT*16][K] row-major (transposed W)
                 const float* __restrict__ bias,      // [NT*16] or nullptr
                 float* __restrict__ C,               // [M][NT*16] row-major
                 int M, int K)
{
    constexpr int NCOL = NT * 16;
    __shared__ __align__(16) _Float16 lA[128 * 32];      // [row][k]
    __shared__ __align__(16) _Float16 lB[NCOL * 32];     // [n][k]

    const int tid  = threadIdx.x;
    const int wave = tid >> 5;
    const int lane = tid & 31;
    const int rowBase = blockIdx.x * 128;

    v8f zero = {};
    v8f acc[NT];
#pragma unroll
    for (int t = 0; t < NT; ++t) acc[t] = zero;

    const int m16   = lane & 15;
    const int khalf = lane >> 4;

    for (int k0 = 0; k0 < K; k0 += 32) {
        // ---- stage A tile (128x32): 4x float4 loads -> 16 f16 -> 2x b128 LDS stores
        {
            int r  = tid >> 1;
            int kk = (tid & 1) * 16;
            int gr = rowBase + r;
            union { _Float16 h[16]; uint4 u[2]; } pk;
            if (gr < M) {
                const float4* sp = (const float4*)(A + (long)gr * strideA + k0 + kk);
#pragma unroll
                for (int q = 0; q < 4; ++q) {
                    float4 f = sp[q];
                    pk.h[q * 4 + 0] = (_Float16)f.x;
                    pk.h[q * 4 + 1] = (_Float16)f.y;
                    pk.h[q * 4 + 2] = (_Float16)f.z;
                    pk.h[q * 4 + 3] = (_Float16)f.w;
                }
            } else {
                pk.u[0] = make_uint4(0, 0, 0, 0);
                pk.u[1] = make_uint4(0, 0, 0, 0);
            }
            uint4* dp = (uint4*)&lA[r * 32 + kk];
            dp[0] = pk.u[0];
            dp[1] = pk.u[1];
        }
        // ---- stage B tile (NCOL x 32): contiguous 16B global -> 16B LDS chunks
        {
            constexpr int CHUNKS = NCOL * 4;            // 8-half chunks per k-step
#pragma unroll
            for (int i = tid; i < CHUNKS; i += 256) {
                int n  = i >> 2;
                int kc = (i & 3) * 8;
                *(uint4*)&lB[n * 32 + kc] =
                    *(const uint4*)&Bt[(long)n * K + k0 + kc];
            }
        }
        __syncthreads();

        // A fragment per ISA 16-bit 16x32 layout:
        //  lanes 0-15 (m): v0-3 -> K 0..7,  v4-7 -> K 16..23
        //  lanes16-31 (m): v0-3 -> K 8..15, v4-7 -> K 24..31
        union { uint4 u[2]; v16h v; } af, bfrag;
        const _Float16* ar = &lA[(wave * 16 + m16) * 32];
        af.u[0] = *(const uint4*)(ar + khalf * 8);
        af.u[1] = *(const uint4*)(ar + 16 + khalf * 8);

#pragma unroll
        for (int t = 0; t < NT; ++t) {
            // B fragment: lane n=lane%16, K = khalf*8..+7 then 16+khalf*8..+7 (same as A)
            const _Float16* br = &lB[(t * 16 + m16) * 32];
            bfrag.u[0] = *(const uint4*)(br + khalf * 8);
            bfrag.u[1] = *(const uint4*)(br + 16 + khalf * 8);
            acc[t] = __builtin_amdgcn_wmma_f32_16x16x32_f16(
                false, af.v, false, bfrag.v, (short)0, acc[t], false, false);
        }
        __syncthreads();
    }

    // store C: element (p,lane): m = p + 8*(lane>>4), n = t*16 + lane%16
    const int mloc = wave * 16 + 8 * khalf;
#pragma unroll
    for (int t = 0; t < NT; ++t) {
        int gn = t * 16 + m16;
        float bv = bias ? bias[gn] : 0.0f;
#pragma unroll
        for (int p = 0; p < 8; ++p) {
            int gm = rowBase + mloc + p;
            if (gm < M) C[(long)gm * NCOL + gn] = acc[t][p] + bv;
        }
    }
}

// ---------------- attention logits: al[n,h] = <H[n,h,:], att[h,:]> ----------------
__global__ void k_attn_logits(const float* __restrict__ H,
                              const float* __restrict__ att_src,
                              const float* __restrict__ att_dst,
                              float* __restrict__ als, float* __restrict__ ald)
{
    int i = blockIdx.x * blockDim.x + threadIdx.x;
    if (i >= N_NODES * 4) return;
    int h = i & 3, n = i >> 2;
    const float* hp = H + (long)n * HC + h * FH;
    const float* as = att_src + h * FH;
    const float* ad = att_dst + h * FH;
    float s = 0.0f, d = 0.0f;
#pragma unroll 8
    for (int c = 0; c < FH; ++c) { float v = hp[c]; s += v * as[c]; d += v * ad[c]; }
    als[i] = s; ald[i] = d;
}

// ---------------- edge pass 1: leaky-relu logits + segment max ----------------
__global__ void k_edge_logit_max(const int* __restrict__ ei,
                                 const float* __restrict__ als, const float* __restrict__ ald,
                                 float* __restrict__ logits, float* __restrict__ maxb)
{
    int i = blockIdx.x * blockDim.x + threadIdx.x;
    if (i >= E_TOT * 4) return;
    int h = i & 3, e = i >> 2, s, d;
    edge_sd(ei, e, s, d);
    float v = als[s * 4 + h] + ald[d * 4 + h];
    v = (v > 0.0f) ? v : v * NEG_SLOPE;
    logits[i] = v;
    atomicMaxFloat(&maxb[d * 4 + h], v);
}

// ---------------- edge pass 2: exp + segment sum ----------------
__global__ void k_edge_exp_sum(const int* __restrict__ ei,
                               const float* __restrict__ maxb,
                               float* __restrict__ logits, float* __restrict__ denom)
{
    int i = blockIdx.x * blockDim.x + threadIdx.x;
    if (i >= E_TOT * 4) return;
    int h = i & 3, e = i >> 2, s, d;
    edge_sd(ei, e, s, d);
    float v = expf(logits[i] - maxb[d * 4 + h]);
    logits[i] = v;
    atomicAdd(&denom[d * 4 + h], v);
}

// ---------------- edge pass 3: weighted scatter of messages ----------------
__global__ __launch_bounds__(256)
void k_edge_scatter(const int* __restrict__ ei,
                    const float* __restrict__ H,       // [N,256] features
                    const float* __restrict__ logits,  // [E_tot,4] exp values
                    const float* __restrict__ denom,   // [N,4]
                    float* __restrict__ out)           // [N,256] accum
{
    int e = blockIdx.x;
    int tid = threadIdx.x;           // tid = h*64 + c
    int h = tid >> 6, s, d;
    edge_sd(ei, e, s, d);
    float alpha = logits[e * 4 + h] / (denom[d * 4 + h] + EPS_DENOM);
    atomicAdd(&out[(long)d * HC + tid], H[(long)s * HC + tid] * alpha);
}

// ---------------- epilogue L1: out = relu(out + b1) in place ----------------
__global__ void k_bias_relu_256(float* __restrict__ buf, const float* __restrict__ bias)
{
    long i = (long)blockIdx.x * blockDim.x + threadIdx.x;
    if (i >= (long)N_NODES * HC) return;
    float v = buf[i] + bias[i & (HC - 1)];
    buf[i] = v > 0.0f ? v : 0.0f;
}

// ---------------- epilogue L2: mean over 4 heads + b2 + relu -> xt[N,64] --------
__global__ void k_heads_mean_relu(const float* __restrict__ buf, const float* __restrict__ b2,
                                  float* __restrict__ xt)
{
    long i = (long)blockIdx.x * blockDim.x + threadIdx.x;
    if (i >= (long)N_NODES * FH) return;
    int n = (int)(i >> 6), c = (int)(i & 63);
    const float* p = buf + (long)n * HC;
    float v = 0.25f * (p[c] + p[64 + c] + p[128 + c] + p[192 + c]) + b2[c];
    xt[i] = v > 0.0f ? v : 0.0f;
}

// ---------------- GRU gate fusion ----------------
__global__ void k_gru_gate(const float* __restrict__ gi, const float* __restrict__ gh,
                           const float* __restrict__ hprev, float* __restrict__ hnew)
{
    long i = (long)blockIdx.x * blockDim.x + threadIdx.x;
    if (i >= (long)N_NODES * FH) return;
    int n = (int)(i >> 6), c = (int)(i & 63);
    const float* gip = gi + (long)n * 192;
    const float* ghp = gh + (long)n * 192;
    float r = sigmoidf_(gip[c]       + ghp[c]);
    float z = sigmoidf_(gip[64 + c]  + ghp[64 + c]);
    float g = tanhf    (gip[128 + c] + r * ghp[128 + c]);
    hnew[i] = (1.0f - z) * g + z * hprev[i];
}

// ---------------- final linear: out[n] = h[n,:] . W_out + b_out ----------------
__global__ void k_final(const float* __restrict__ h, const float* __restrict__ Wout,
                        const float* __restrict__ bout, float* __restrict__ out)
{
    int n = blockIdx.x * blockDim.x + threadIdx.x;
    if (n >= N_NODES) return;
    const float* hp = h + (long)n * FH;
    float s = bout[0];
#pragma unroll 8
    for (int c = 0; c < FH; ++c) s += hp[c] * Wout[c];
    out[n] = s;
}

// ---------------- host side ----------------
static inline size_t alignup(size_t x) { return (x + 255) & ~(size_t)255; }
static inline int cdiv(long a, int b) { return (int)((a + b - 1) / b); }

extern "C" void kernel_launch(void* const* d_in, const int* in_sizes, int n_in,
                              void* d_out, int out_size, void* d_ws, size_t ws_size,
                              hipStream_t stream)
{
    const float* x_seq  = (const float*)d_in[0];   // [N,T,128]
    const int*   ei_all = (const int*)  d_in[1];   // [T,2,E]
    const float* W1     = (const float*)d_in[2];
    const float* asrc1  = (const float*)d_in[3];
    const float* adst1  = (const float*)d_in[4];
    const float* b1     = (const float*)d_in[5];
    const float* W2     = (const float*)d_in[6];
    const float* asrc2  = (const float*)d_in[7];
    const float* adst2  = (const float*)d_in[8];
    const float* b2     = (const float*)d_in[9];
    const float* W_ih   = (const float*)d_in[10];
    const float* W_hh   = (const float*)d_in[11];
    const float* b_ih   = (const float*)d_in[12];
    const float* b_hh   = (const float*)d_in[13];
    const float* W_out  = (const float*)d_in[14];
    const float* b_out  = (const float*)d_in[15];
    float* out = (float*)d_out;

    // workspace carve-out
    char* ws = (char*)d_ws;
    size_t off = 0;
    auto alloc = [&](size_t bytes) -> void* { void* p = ws + off; off = alignup(off + bytes); return p; };
    _Float16* W1t  = (_Float16*)alloc((size_t)F_INP * HC * 2);   // [256][128]
    _Float16* W2t  = (_Float16*)alloc((size_t)HC * HC * 2);      // [256][256]
    _Float16* Wiht = (_Float16*)alloc((size_t)FH * 192 * 2);     // [192][64]
    _Float16* Whht = (_Float16*)alloc((size_t)FH * 192 * 2);     // [192][64]
    float* bufA   = (float*)alloc((size_t)N_NODES * HC * 4);     // GEMM outputs / features
    float* bufB   = (float*)alloc((size_t)N_NODES * HC * 4);     // scatter accumulators
    float* als    = (float*)alloc((size_t)N_NODES * 4 * 4);
    float* ald    = (float*)alloc((size_t)N_NODES * 4 * 4);
    float* maxb   = (float*)alloc((size_t)N_NODES * 4 * 4);
    float* denom  = (float*)alloc((size_t)N_NODES * 4 * 4);
    float* logits = (float*)alloc((size_t)E_TOT * 4 * 4);
    float* xt     = (float*)alloc((size_t)N_NODES * FH * 4);
    float* gi     = (float*)alloc((size_t)N_NODES * 192 * 4);
    float* gh     = (float*)alloc((size_t)N_NODES * 192 * 4);
    float* hA     = (float*)alloc((size_t)N_NODES * FH * 4);
    float* hB     = (float*)alloc((size_t)N_NODES * FH * 4);
    (void)ws_size; (void)n_in; (void)in_sizes; (void)out_size;

    const float NEG_INF = -__builtin_inff();

    // convert weights to f16, transposed to [N][K] (idempotent each call)
    k_cvt_f16_T<<<cdiv(F_INP * HC, 256), 256, 0, stream>>>(W1, W1t, F_INP, HC);
    k_cvt_f16_T<<<cdiv(HC * HC, 256), 256, 0, stream>>>(W2, W2t, HC, HC);
    k_cvt_f16_T<<<cdiv(FH * 192, 256), 256, 0, stream>>>(W_ih, Wiht, FH, 192);
    k_cvt_f16_T<<<cdiv(FH * 192, 256), 256, 0, stream>>>(W_hh, Whht, FH, 192);

    // h0 = 0
    k_fill_f32<<<cdiv((long)N_NODES * FH, 256), 256, 0, stream>>>(hA, 0.0f, (long)N_NODES * FH);

    float* hcur = hA;
    float* hnxt = hB;

    const int gemmBlocks = cdiv(N_NODES, 128);
    const int nhBlocks   = cdiv(N_NODES * 4, 256);
    const int ehBlocks   = cdiv((long)E_TOT * 4, 256);
    const int n256Blocks = cdiv((long)N_NODES * HC, 256);
    const int n64Blocks  = cdiv((long)N_NODES * FH, 256);

    for (int t = 0; t < T_STEPS; ++t) {
        const int* ei_t = ei_all + (long)t * 2 * E_EDGES;
        const float* x_t = x_seq + (long)t * F_INP;          // row stride T*F_IN

        // ---- GAT layer 1 ----
        k_gemm_wmma<16><<<gemmBlocks, 256, 0, stream>>>(
            x_t, (long)T_STEPS * F_INP, W1t, nullptr, bufA, N_NODES, F_INP);
        k_attn_logits<<<nhBlocks, 256, 0, stream>>>(bufA, asrc1, adst1, als, ald);
        k_fill_f32<<<nhBlocks, 256, 0, stream>>>(maxb, NEG_INF, (long)N_NODES * 4);
        k_fill_f32<<<nhBlocks, 256, 0, stream>>>(denom, 0.0f, (long)N_NODES * 4);
        k_fill_f32<<<n256Blocks, 256, 0, stream>>>(bufB, 0.0f, (long)N_NODES * HC);
        k_edge_logit_max<<<ehBlocks, 256, 0, stream>>>(ei_t, als, ald, logits, maxb);
        k_edge_exp_sum<<<ehBlocks, 256, 0, stream>>>(ei_t, maxb, logits, denom);
        k_edge_scatter<<<E_TOT, 256, 0, stream>>>(ei_t, bufA, logits, denom, bufB);
        k_bias_relu_256<<<n256Blocks, 256, 0, stream>>>(bufB, b1);   // bufB = h1 [N,256]

        // ---- GAT layer 2 ----
        k_gemm_wmma<16><<<gemmBlocks, 256, 0, stream>>>(
            bufB, (long)HC, W2t, nullptr, bufA, N_NODES, HC);        // bufA = h2 feats
        k_attn_logits<<<nhBlocks, 256, 0, stream>>>(bufA, asrc2, adst2, als, ald);
        k_fill_f32<<<nhBlocks, 256, 0, stream>>>(maxb, NEG_INF, (long)N_NODES * 4);
        k_fill_f32<<<nhBlocks, 256, 0, stream>>>(denom, 0.0f, (long)N_NODES * 4);
        k_fill_f32<<<n256Blocks, 256, 0, stream>>>(bufB, 0.0f, (long)N_NODES * HC);
        k_edge_logit_max<<<ehBlocks, 256, 0, stream>>>(ei_t, als, ald, logits, maxb);
        k_edge_exp_sum<<<ehBlocks, 256, 0, stream>>>(ei_t, maxb, logits, denom);
        k_edge_scatter<<<E_TOT, 256, 0, stream>>>(ei_t, bufA, logits, denom, bufB);
        k_heads_mean_relu<<<n64Blocks, 256, 0, stream>>>(bufB, b2, xt); // xt [N,64]

        // ---- GRU step ----
        k_gemm_wmma<12><<<gemmBlocks, 256, 0, stream>>>(
            xt, (long)FH, Wiht, b_ih, gi, N_NODES, FH);
        k_gemm_wmma<12><<<gemmBlocks, 256, 0, stream>>>(
            hcur, (long)FH, Whht, b_hh, gh, N_NODES, FH);
        k_gru_gate<<<n64Blocks, 256, 0, stream>>>(gi, gh, hcur, hnxt);
        float* tmp = hcur; hcur = hnxt; hnxt = tmp;
    }

    // ---- final linear ----
    k_final<<<cdiv(N_NODES, 256), 256, 0, stream>>>(hcur, W_out, b_out, out);
}
